// span_embed_layer_66640712564949
// MI455X (gfx1250) — matrix operand order repaired and checked by reading, not compiled, not run on hardware
//
#include <hip/hip_runtime.h>
#include <hip/hip_bf16.h>

typedef unsigned short u16;
typedef __attribute__((ext_vector_type(16))) __bf16 v16bf;
typedef __attribute__((ext_vector_type(8)))  float  v8f;

// Problem constants (match reference)
#define BB   8
#define NN   2048
#define MM   1024
#define HH   1024
#define DIS  20
#define KC   3092           // 3*H + DIS
#define KPAD 3104           // 97 * 32
#define KSTEPS 97
#define NTILES 64           // HH / 16        (16-wide col tiles, for packing)
#define MT2  256            // (BB*MM) / 32   (32-row supertiles)
#define NT2  16             // HH / 64        (64-col supertiles)

// halves between 16-row A groups / between adjacent packed B tiles (both 49664)
#define AROWSTRIDE (16 * KPAD)        // 49664 halves = 99328 B
#define BTILESTRIDE (KSTEPS * 32 * 16) // 49664 halves = 99328 B

__device__ __forceinline__ u16 f2bf(float f) {
  unsigned u = __float_as_uint(f);
  unsigned r = u + 0x7FFFu + ((u >> 16) & 1u);   // RNE
  return (u16)(r >> 16);
}

// ---------------------------------------------------------------------------
// Kernel 0: pack out_w [KC x HH] f32 row-major into per-lane bf16 B-fragments
// Bpack layout: [tn][k][lane][j] halves, j=0..15, Kidx = 8*(lane>>4) + j (j<8), 16 + 8*(lane>>4) + (j-8) (j>=8),
// col = tn*16 + (lane&15).  Rows K >= KC are zero-padded.
// Total halves = NTILES * KSTEPS * 32 * 16 = 3,178,496 = 12416 * 256.
// ---------------------------------------------------------------------------
__global__ void pack_w_kernel(const float* __restrict__ out_w,
                              u16* __restrict__ Bpack) {
  unsigned idx = blockIdx.x * 256u + threadIdx.x;
  unsigned j    = idx & 15u;
  unsigned lane = (idx >> 4) & 31u;
  unsigned t    = idx >> 9;          // (tn*97 + k)
  unsigned k    = t % KSTEPS;
  unsigned tn   = t / KSTEPS;
  unsigned K = k * 32u + ((lane >> 4) * 8u) + (j < 8u ? j : 16u + (j - 8u));
  unsigned Ncol = tn * 16u + (lane & 15u);
  float v = (K < KC) ? out_w[(size_t)K * HH + Ncol] : 0.0f;
  Bpack[idx] = f2bf(v);
}

// ---------------------------------------------------------------------------
// Kernel 1: per-token scalar score = ctx . scale_w + scale_b   [B*N]
// One wave (32 lanes) per token row; 8 waves per 256-thread block.
// ---------------------------------------------------------------------------
__global__ void scores_kernel(const float* __restrict__ ctx,
                              const float* __restrict__ scale_w,
                              const float* __restrict__ scale_b,
                              float* __restrict__ scores) {
  int lane = threadIdx.x & 31;
  int wid  = threadIdx.x >> 5;
  int row  = blockIdx.x * 8 + wid;           // 0 .. B*N-1
  const float* x = ctx + (size_t)row * HH;
  float s = 0.f;
  #pragma unroll
  for (int i = 0; i < HH / 32; ++i) s += x[lane + i * 32] * scale_w[lane + i * 32];
  #pragma unroll
  for (int off = 16; off; off >>= 1) s += __shfl_xor(s, off, 32);
  if (lane == 0) scores[row] = s + scale_b[0];
}

// ---------------------------------------------------------------------------
// Kernel 2: one block per span. Masked softmax over the <=63 span tokens,
// weighted-sum embedding, start/end gathers, length-bucket embedding.
// Writes one bf16 feature row of KPAD halves:
//   [0:20) len_embed | [20:1044) embed | [1044:2068) start | [2068:3092) end | 12 zeros
// ---------------------------------------------------------------------------
__global__ void feat_kernel(const float* __restrict__ ctx,
                            const int*   __restrict__ info,      // [B,M,4] int32
                            const float* __restrict__ len_embed, // [20, DIS]
                            const float* __restrict__ scores,    // [B, N]
                            u16* __restrict__ featA) {
  int span = blockIdx.x;             // b*M + m
  int b = span >> 10;                // M = 1024
  __shared__ float sw[64];
  __shared__ int   ss[4];            // start, end, bucket, len

  if (threadIdx.x == 0) {
    int s = info[(size_t)span * 4 + 2];
    int e = info[(size_t)span * 4 + 3];
    int len = e - s;                               // in [1, 63]
    int bucket = 32 - __clz(len);                  // floor(log2(len)) + 1
    if (bucket > 9) bucket = 9;
    ss[0] = s; ss[1] = e; ss[2] = bucket; ss[3] = len;
    const float* sc = scores + (size_t)b * NN + s;
    float mx = -1e30f;
    for (int t = 0; t < len; ++t) mx = fmaxf(mx, sc[t]);
    float sum = 0.f;
    for (int t = 0; t < len; ++t) { float ev = expf(sc[t] - mx); sw[t] = ev; sum += ev; }
    float inv = 1.f / sum;
    for (int t = 0; t < len; ++t) sw[t] *= inv;
  }
  __syncthreads();

  int s = ss[0], e = ss[1], bucket = ss[2], len = ss[3];
  int tid = threadIdx.x;

  const float* base = ctx + ((size_t)b * NN + s) * HH;   // start row
  float acc0 = 0.f, acc1 = 0.f, acc2 = 0.f, acc3 = 0.f;
  for (int t = 0; t < len; ++t) {
    float w = sw[t];
    const float* r = base + (size_t)t * HH;
    acc0 += w * r[tid];
    acc1 += w * r[tid + 256];
    acc2 += w * r[tid + 512];
    acc3 += w * r[tid + 768];
  }

  u16* frow = featA + (size_t)span * KPAD;
  const float* rs = base;
  const float* re = ctx + ((size_t)b * NN + e) * HH;
  frow[DIS + tid      ] = f2bf(acc0);
  frow[DIS + tid + 256] = f2bf(acc1);
  frow[DIS + tid + 512] = f2bf(acc2);
  frow[DIS + tid + 768] = f2bf(acc3);
  #pragma unroll
  for (int k2 = 0; k2 < 4; ++k2) {
    int h = tid + k2 * 256;
    frow[DIS + HH     + h] = f2bf(rs[h]);
    frow[DIS + 2 * HH + h] = f2bf(re[h]);
  }
  if (tid < DIS) frow[tid] = f2bf(len_embed[bucket * DIS + tid]);
  if (tid < KPAD - KC) frow[KC + tid] = 0;   // zero K padding
}

// ---------------------------------------------------------------------------
// Kernel 3: bf16 WMMA GEMM  [8192 x KPAD] x [KPAD x 1024] -> f32, +bias, ReLU.
// Register-blocked: one wave computes a 32x64 output tile = 2x4 WMMA tiles.
// Per K-step: 2 A fragments + 4 B fragments -> 8 v_wmma_f32_16x16x32_bf16.
// Only TWO base pointers; all fragment addresses are immediate offsets so the
// backend emits  global_load_b128 v[..], v[base], off offset:imm  with no
// per-fragment 64-bit address math.  launch_bounds(256,1) unlocks the full
// VGPR budget so the 64 accumulator regs never spill.
// ---------------------------------------------------------------------------
__global__ void __launch_bounds__(256, 1)
gemm_kernel(const u16* __restrict__ featA,
            const u16* __restrict__ Bpack,
            const float* __restrict__ bias,
            float* __restrict__ out) {
  int lane = threadIdx.x & 31;
  int wid  = threadIdx.x >> 5;
  int tile = blockIdx.x * 8 + wid;       // 0 .. 4095
  int tm2 = tile >> 4;                   // 0 .. 255  (32-row supertile)
  int tn2 = tile & 15;                   // 0 .. 15   (64-col supertile)
  int m  = lane & 15;
  int hi = lane >> 4;

  // A: lane hi half selects K = 8*hi + [0..7] and K = 16 + 8*hi + [0..7]
  const u16* abase = featA + (size_t)(tm2 * 32 + m) * KPAD + 8 * hi;
  // B: packed fragments, 16 contiguous halves per lane per k-step
  const u16* bbase = Bpack + ((size_t)(tn2 * 4) * KSTEPS * 32 + lane) * 16;

  v8f c[2][4];
  #pragma unroll
  for (int i = 0; i < 2; ++i)
    #pragma unroll
    for (int j = 0; j < 4; ++j) c[i][j] = {};

  for (int k = 0; k < KSTEPS; ++k) {
    union F { uint4 q[2]; v16bf v; };
    F a[2], bf[4];
    #pragma unroll
    for (int i = 0; i < 2; ++i) {
      a[i].q[0] = *reinterpret_cast<const uint4*>(abase + i * AROWSTRIDE + k * 32);
      a[i].q[1] = *reinterpret_cast<const uint4*>(abase + i * AROWSTRIDE + k * 32 + 16);
    }
    #pragma unroll
    for (int j = 0; j < 4; ++j) {
      bf[j].q[0] = *reinterpret_cast<const uint4*>(bbase + j * BTILESTRIDE + k * 512);
      bf[j].q[1] = *reinterpret_cast<const uint4*>(bbase + j * BTILESTRIDE + k * 512 + 8);
    }
    #pragma unroll
    for (int i = 0; i < 2; ++i)
      #pragma unroll
      for (int j = 0; j < 4; ++j)
        c[i][j] = __builtin_amdgcn_wmma_f32_16x16x32_bf16(
            /*neg_a=*/false, a[i].v, /*neg_b=*/false, bf[j].v,
            /*c_mod=*/(short)0, c[i][j], /*reuse_a=*/false, /*reuse_b=*/false);
  }

  #pragma unroll
  for (int j = 0; j < 4; ++j) {
    int col  = (tn2 * 4 + j) * 16 + m;
    float bv = bias[col];
    #pragma unroll
    for (int i = 0; i < 2; ++i) {
      int row0 = tm2 * 32 + i * 16 + 8 * hi;
      #pragma unroll
      for (int r = 0; r < 8; ++r) {
        float v = c[i][j][r] + bv;
        __builtin_nontemporal_store(v > 0.f ? v : 0.f,
                                    out + (size_t)(row0 + r) * HH + col);
      }
    }
  }
}

// ---------------------------------------------------------------------------
extern "C" void kernel_launch(void* const* d_in, const int* in_sizes, int n_in,
                              void* d_out, int out_size, void* d_ws, size_t ws_size,
                              hipStream_t stream) {
  const float* ctx       = (const float*)d_in[0];   // [B,N,H]
  const int*   info      = (const int*)  d_in[1];   // [B,M,4] int32 (JAX x64 off)
  const float* len_embed = (const float*)d_in[2];   // [20,DIS]
  const float* scale_w   = (const float*)d_in[3];   // [H,1]
  const float* scale_b   = (const float*)d_in[4];   // [1]
  const float* out_w     = (const float*)d_in[5];   // [KC,H]
  const float* out_b     = (const float*)d_in[6];   // [H]
  float* out = (float*)d_out;                       // [B,M,H]

  char* ws = (char*)d_ws;
  float* scores = (float*)(ws);                               // 64 KB
  u16*   Bpack  = (u16*)  (ws + 65536);                       // ~6.36 MB
  u16*   featA  = (u16*)  (ws + 6422528);                     // ~50.9 MB

  // 1) pack out_w into bf16 B fragments (zero-padded to KPAD)
  pack_w_kernel<<<12416, 256, 0, stream>>>(out_w, Bpack);
  // 2) per-token scores
  scores_kernel<<<(BB * NN) / 8, 256, 0, stream>>>(ctx, scale_w, scale_b, scores);
  // 3) per-span softmax + feature rows (bf16)
  feat_kernel<<<BB * MM, 256, 0, stream>>>(ctx, info, len_embed, scores, featA);
  // 4) register-blocked WMMA GEMM + bias + ReLU
  gemm_kernel<<<(MT2 * NT2) / 8, 256, 0, stream>>>(featA, Bpack, out_b, out);
}
